// NormMSVectorQuantizer_69733089017858
// MI455X (gfx1250) — matrix operand order repaired and checked by reading, not compiled, not run on hardware
//
#include <hip/hip_runtime.h>
#include <hip/hip_bf16.h>
#include <stdint.h>

typedef __attribute__((ext_vector_type(16))) __bf16 v16bf;
typedef __attribute__((ext_vector_type(8)))  __bf16 v8bf;
typedef __attribute__((ext_vector_type(8)))  float  v8f;

#define B_    64
#define T_    16
#define D_    256
#define V_    8192
#define C_    64
#define L_    (B_ * T_)            // 1024
#define NPOS  (L_ * C_)            // 65536 positions (l, c)
#define ZQ_ELEMS   (B_ * 1024 * D_)  // 16777216
#define IDX_PER_L  85                // 1+4+16+64
#define NTILES_EB  (V_ / 16)         // 512
#define KTILES_EB  (D_ / 32)         // 8
#define KTILES_W   (3 * D_ / 32)     // 24
#define NTILES_W   (D_ / 16)         // 16

// ---------------- fragment index maps (CDNA5 16-bit WMMA layouts) ----------------
// A matrix 16x32 bf16: lane M = lane&15 ; K base kb = 0 (lanes<16) / 8 (lanes>=16)
// VGPRs 0..3 hold K = kb..kb+7 (consecutive), VGPRs 4..7 hold K = kb+16..kb+23.
// => one fragment = two 16B-aligned contiguous 8x bf16 loads.
__device__ __forceinline__ v16bf load_a_frag(const __bf16* rowp, int kb) {
  v8bf lo = *(const v8bf*)(rowp + kb);
  v8bf hi = *(const v8bf*)(rowp + kb + 16);
  return __builtin_shufflevector(lo, hi, 0, 1, 2, 3, 4, 5, 6, 7,
                                 8, 9, 10, 11, 12, 13, 14, 15);
}
// B matrix 32x16 bf16: N = lane&15 ; K = (e<8?0:16) + (lane<16?0:8) + (e&7)
__device__ __forceinline__ void b_kn_of(int lane, int e, int& K, int& N) {
  K = ((e < 8) ? 0 : 16) + ((lane < 16) ? 0 : 8) + (e & 7);
  N = lane & 15;
}

__device__ __forceinline__ float block_reduce_sum_256(float v, float* sred8) {
#pragma unroll
  for (int off = 16; off > 0; off >>= 1) v += __shfl_down(v, off, 32);
  const int lane = threadIdx.x & 31, w = threadIdx.x >> 5;
  if (lane == 0) sred8[w] = v;
  __syncthreads();
  if (w == 0) {
    float r = (lane < 8) ? sred8[lane] : 0.0f;
#pragma unroll
    for (int off = 4; off > 0; off >>= 1) r += __shfl_down(r, off, 32);
    if (lane == 0) sred8[0] = r;
  }
  __syncthreads();
  float r = sred8[0];
  __syncthreads();
  return r;
}

// ---------------- preprocessing kernels ----------------
__global__ void k_norm_embed(const float* __restrict__ emb, float* __restrict__ en) {
  __shared__ float sred[8];
  const int v = blockIdx.x, d = threadIdx.x;
  float x = emb[(size_t)v * D_ + d];
  float ss = block_reduce_sum_256(x * x, sred);
  float nrm = fmaxf(sqrtf(ss), 1e-12f);
  en[(size_t)v * D_ + d] = x / nrm;
}

__global__ void k_pack_eb(const float* __restrict__ en, __bf16* __restrict__ ebfrag) {
  const int nt = blockIdx.x >> 3, kt = blockIdx.x & 7;
  for (int f = threadIdx.x; f < 512; f += 256) {
    int lane = f >> 4, e = f & 15, K, N;
    b_kn_of(lane, e, K, N);
    float v = en[(size_t)(nt * 16 + N) * D_ + kt * 32 + K];
    ebfrag[(size_t)blockIdx.x * 512 + f] = (__bf16)v;
  }
}

__global__ void k_pack_w(const float* __restrict__ phi_w, __bf16* __restrict__ wfrag) {
  const int nt = blockIdx.x / KTILES_W, kt = blockIdx.x % KTILES_W;
  for (int f = threadIdx.x; f < 512; f += 256) {
    int lane = f >> 4, e = f & 15, K, N;
    b_kn_of(lane, e, K, N);
    int kk = kt * 32 + K;           // K-dim index of im2col GEMM
    int tap = kk >> 8, i = kk & 255, o = nt * 16 + N;
    float v = phi_w[((size_t)o * D_ + i) * 3 + tap];
    wfrag[(size_t)blockIdx.x * 512 + f] = (__bf16)v;
  }
}

__global__ void k_norm_scatter(const float* __restrict__ z, const int* __restrict__ chans,
                               float* __restrict__ zLND, float* __restrict__ frest) {
  __shared__ float sred[8];
  const int row = blockIdx.x;            // b*1024 + n
  const int b = row >> 10, n = row & 1023, t = n >> 6, c = n & 63, d = threadIdx.x;
  float v = z[(size_t)row * D_ + d];
  float ss = block_reduce_sum_256(v * v, sred);
  v /= fmaxf(sqrtf(ss), 1e-12f);
  int ch = chans[b * 1024 + c];          // permutation, first 64 entries per batch
  size_t dst = ((size_t)((b * T_ + t) * C_ + ch)) * D_ + d;
  zLND[dst] = v;
  frest[dst] = v;
}

// ---------------- per-stage kernels ----------------
__global__ void k_group_mean_pack(const float* __restrict__ frest, __bf16* __restrict__ qbuf,
                                  int g, int sub) {
  __shared__ float sred[8];
  const int r = blockIdx.x, d = threadIdx.x;
  const int l = r / g, gi = r % g;
  size_t base = ((size_t)l * C_ + gi * sub) * D_ + d;
  float s = 0.0f;
  for (int j = 0; j < sub; ++j) s += frest[base + (size_t)j * D_];
  s /= (float)sub;
  float ss = block_reduce_sum_256(s * s, sred);
  float q = s / fmaxf(sqrtf(ss), 1e-12f);
  qbuf[(size_t)r * D_ + d] = (__bf16)q;
}

// Codebook scoring + argmax: (32 rows) x (V=8192) via v_wmma_f32_16x16x32_bf16.
// Two resident A row-tiles x two interleaved N-streams: every B fragment pair
// feeds 4 WMMAs, so each exposed load latency is amortized 4x and total B
// traffic from L2 is halved vs a 16-row block.
__global__ void __launch_bounds__(256, 1)
k_argmax(const __bf16* __restrict__ qbuf, const __bf16* __restrict__ ebfrag,
         int* __restrict__ idxbuf, float* __restrict__ counts,
         float* __restrict__ outIdx, int g, int stageOff) {
  __shared__ __bf16 qs[32 * D_];
  __shared__ float smax[8][32];
  __shared__ int   sidx[8][32];
  const int tid = threadIdx.x, lane = tid & 31, w = tid >> 5;
  const size_t rowbase = (size_t)blockIdx.x * 32;

  {
    uint4* dst = (uint4*)qs;
    const uint4* src = (const uint4*)(qbuf + rowbase * D_);
    for (int i = tid; i < 32 * D_ / 8; i += 256) dst[i] = src[i];
  }
  __syncthreads();

  const int M = lane & 15;
  const int kb = (lane < 16) ? 0 : 8;
  v16bf af[2][8];
#pragma unroll
  for (int t = 0; t < 2; ++t)
#pragma unroll
    for (int kt = 0; kt < 8; ++kt)
      af[t][kt] = load_a_frag(qs + (t * 16 + M) * D_ + kt * 32, kb);

  float best[2][8];
  int bidx[2][8];
#pragma unroll
  for (int t = 0; t < 2; ++t)
#pragma unroll
    for (int r = 0; r < 8; ++r) { best[t][r] = -3.0e38f; bidx[t][r] = 0; }

  const char* bbase = (const char*)ebfrag + (size_t)lane * 32;
  auto ldb = [&](int nt, int kt) -> v16bf {
    return *(const v16bf*)(bbase + ((size_t)(nt * 8 + kt)) * 1024);
  };

  for (int nt = w; nt < NTILES_EB / 2; nt += 8) {
    const int nt2 = nt + NTILES_EB / 2;
    v8f acc00 = {0.f, 0.f, 0.f, 0.f, 0.f, 0.f, 0.f, 0.f};   // tile0, stream0
    v8f acc01 = {0.f, 0.f, 0.f, 0.f, 0.f, 0.f, 0.f, 0.f};   // tile1, stream0
    v8f acc10 = {0.f, 0.f, 0.f, 0.f, 0.f, 0.f, 0.f, 0.f};   // tile0, stream1
    v8f acc11 = {0.f, 0.f, 0.f, 0.f, 0.f, 0.f, 0.f, 0.f};   // tile1, stream1
#pragma unroll
    for (int kt = 0; kt < 8; ++kt) {
      v16bf bf0 = ldb(nt, kt);
      v16bf bf1 = ldb(nt2, kt);
      acc00 = __builtin_amdgcn_wmma_f32_16x16x32_bf16(false, af[0][kt], false, bf0,
                                                      (short)0, acc00, false, false);
      acc01 = __builtin_amdgcn_wmma_f32_16x16x32_bf16(false, af[1][kt], false, bf0,
                                                      (short)0, acc01, false, false);
      acc10 = __builtin_amdgcn_wmma_f32_16x16x32_bf16(false, af[0][kt], false, bf1,
                                                      (short)0, acc10, false, false);
      acc11 = __builtin_amdgcn_wmma_f32_16x16x32_bf16(false, af[1][kt], false, bf1,
                                                      (short)0, acc11, false, false);
    }
    const int col0 = nt * 16 + (lane & 15);
    const int col1 = nt2 * 16 + (lane & 15);
#pragma unroll
    for (int r = 0; r < 8; ++r) {
      if (acc00[r] > best[0][r]) { best[0][r] = acc00[r]; bidx[0][r] = col0; }
      if (acc10[r] > best[0][r]) { best[0][r] = acc10[r]; bidx[0][r] = col1; }
      if (acc01[r] > best[1][r]) { best[1][r] = acc01[r]; bidx[1][r] = col0; }
      if (acc11[r] > best[1][r]) { best[1][r] = acc11[r]; bidx[1][r] = col1; }
    }
  }

  // reduce across the 16 lanes sharing the same C-rows (xor stays within the half)
#pragma unroll
  for (int t = 0; t < 2; ++t)
#pragma unroll
    for (int r = 0; r < 8; ++r) {
      float v = best[t][r]; int id = bidx[t][r];
#pragma unroll
      for (int off = 8; off > 0; off >>= 1) {
        float ov = __shfl_xor(v, off, 32);
        int   oi = __shfl_xor(id, off, 32);
        if (ov > v) { v = ov; id = oi; }
      }
      best[t][r] = v; bidx[t][r] = id;
    }
  if ((lane & 15) == 0) {
    int mb = (lane < 16) ? 0 : 8;
#pragma unroll
    for (int t = 0; t < 2; ++t)
#pragma unroll
      for (int r = 0; r < 8; ++r) {
        smax[w][t * 16 + mb + r] = best[t][r];
        sidx[w][t * 16 + mb + r] = bidx[t][r];
      }
  }
  __syncthreads();
  if (tid < 32) {
    float v = smax[0][tid]; int id = sidx[0][tid];
    for (int ww = 1; ww < 8; ++ww) {
      float ov = smax[ww][tid]; int oi = sidx[ww][tid];
      if (ov > v) { v = ov; id = oi; }
    }
    size_t row = rowbase + tid;
    idxbuf[row] = id;
    int l = (int)(row / (size_t)g), gi = (int)(row % (size_t)g);
    outIdx[(size_t)l * IDX_PER_L + stageOff + gi] = (float)id;
    atomicAdd(&counts[id], 1.0f);
  }
}

__global__ void k_build_h(const int* __restrict__ idxbuf, const float* __restrict__ en,
                          float* __restrict__ hbuf, int g, int sub) {
  const int p = blockIdx.x;                 // l*64 + c
  const int l = p >> 6, c = p & 63, d = threadIdx.x;
  int id = idxbuf[(size_t)l * g + c / sub];
  hbuf[(size_t)p * D_ + d] = en[(size_t)id * D_ + d];
}

// conv1d(k=3, SAME) over the 64 channel slots as im2col GEMM (K=768), fused residual.
// Each wave owns one M-tile; two N-streams share each A fragment; B fragments are
// software-pipelined one K-tile ahead (alternating c/d buffers, no register moves).
__global__ void k_conv_update(const float* __restrict__ hbuf, const __bf16* __restrict__ wfrag,
                              const float* __restrict__ phi_b, const float* __restrict__ zLND,
                              float* __restrict__ fhat, float* __restrict__ frest,
                              float* __restrict__ lossAcc) {
  __shared__ __bf16 hs[34 * D_];            // 32 positions + 2 halo rows, bf16
  __shared__ float sred[8];
  const int tid = threadIdx.x, lane = tid & 31, w = tid >> 5;
  const int l = blockIdx.x >> 1, c0 = (blockIdx.x & 1) * 32;
  const size_t base = (size_t)l * C_ * D_;

  // stage halo tile: 4 features per thread-iteration, b64 LDS stores
  for (int i = tid; i < 34 * D_ / 4; i += 256) {
    int flat = i * 4;
    int gr = c0 - 1 + (flat >> 8);          // global channel slot, -1..64
    union { __bf16 b[4]; uint2 u; } pk;
    if (gr >= 0 && gr < C_) {
      float4 f = *(const float4*)(hbuf + base + (size_t)gr * D_ + (flat & 255));
      pk.b[0] = (__bf16)f.x; pk.b[1] = (__bf16)f.y;
      pk.b[2] = (__bf16)f.z; pk.b[3] = (__bf16)f.w;
    } else {
      pk.u = make_uint2(0u, 0u);
    }
    *(uint2*)(hs + flat) = pk.u;
  }
  __syncthreads();

  const int M = lane & 15;
  const int kb = (lane < 16) ? 0 : 8;
  const int mt = w & 1, mbase = mt * 16;    // wave's M-tile
  const int ntb = w >> 1;                   // 0..3
  const char* wbase = (const char*)wfrag + (size_t)lane * 32;
  auto ldw = [&](int nt, int kt) -> v16bf {
    return *(const v16bf*)(wbase + ((size_t)(nt * KTILES_W + kt)) * 1024);
  };
  auto lda = [&](int kt) -> v16bf {
    int tap = kt >> 3, kof = (kt & 7) * 32;
    return load_a_frag(hs + (mbase + M + tap) * D_ + kof, kb);
  };
  float sq = 0.0f;

  for (int pair = 0; pair < 2; ++pair) {
    const int nt0 = ntb + 8 * pair;
    const int nt1 = nt0 + 4;
    v8f acc0 = {0.f, 0.f, 0.f, 0.f, 0.f, 0.f, 0.f, 0.f};
    v8f acc1 = {0.f, 0.f, 0.f, 0.f, 0.f, 0.f, 0.f, 0.f};
    v16bf c0f = ldw(nt0, 0), c1f = ldw(nt1, 0);
    for (int kt = 0; kt < KTILES_W; kt += 2) {
      v16bf d0f = ldw(nt0, kt + 1), d1f = ldw(nt1, kt + 1);
      v16bf af = lda(kt);
      acc0 = __builtin_amdgcn_wmma_f32_16x16x32_bf16(false, af, false, c0f,
                                                     (short)0, acc0, false, false);
      acc1 = __builtin_amdgcn_wmma_f32_16x16x32_bf16(false, af, false, c1f,
                                                     (short)0, acc1, false, false);
      int ktn = (kt + 2 < KTILES_W) ? kt + 2 : 0;
      c0f = ldw(nt0, ktn); c1f = ldw(nt1, ktn);
      af = lda(kt + 1);
      acc0 = __builtin_amdgcn_wmma_f32_16x16x32_bf16(false, af, false, d0f,
                                                     (short)0, acc0, false, false);
      acc1 = __builtin_amdgcn_wmma_f32_16x16x32_bf16(false, af, false, d1f,
                                                     (short)0, acc1, false, false);
    }
#pragma unroll
    for (int s = 0; s < 2; ++s) {
      const v8f& acc = s ? acc1 : acc0;
      int o = (s ? nt1 : nt0) * 16 + (lane & 15);
      float bias = phi_b[o];
#pragma unroll
      for (int r = 0; r < 8; ++r) {
        int m = c0 + mbase + ((lane < 16) ? r : r + 8);
        size_t p = base + (size_t)m * D_ + o;
        float y  = acc[r] + bias;
        float hp = hbuf[p];
        float h  = 0.5f * hp + 0.5f * y;    // RESI = 0.5
        float fh = fhat[p] + h;
        fhat[p]  = fh;
        frest[p] -= h;
        float d = fh - zLND[p];
        sq += d * d;
      }
    }
  }
  float tot = block_reduce_sum_256(sq, sred);
  if (tid == 0) atomicAdd(lossAcc, tot);
}

// ---------------- finalize kernels ----------------
__global__ void k_gather(const float* __restrict__ fhat, const int* __restrict__ chans,
                         float* __restrict__ outZ) {
  const int row = blockIdx.x;               // b*1024 + n
  const int b = row >> 10, n = row & 1023, t = n >> 6, c = n & 63, d = threadIdx.x;
  int ch = chans[b * 1024 + c];
  outZ[(size_t)row * D_ + d] = fhat[((size_t)((b * T_ + t) * C_ + ch)) * D_ + d];
}

__global__ void k_cluster(const float* __restrict__ csIn, const float* __restrict__ counts,
                          float* __restrict__ out) {
  int v = blockIdx.x * 256 + threadIdx.x;
  out[v] = 0.99f * csIn[v] + 0.01f * counts[v];
}

__global__ void k_loss(const float* __restrict__ lossAcc, float* __restrict__ out) {
  // per stage: (1 + BETA) * mean(sq); summed over 4 stages then /4
  out[0] = 1.25f * lossAcc[0] / (4.0f * (float)ZQ_ELEMS);
}

// ---------------- host launcher ----------------
extern "C" void kernel_launch(void* const* d_in, const int* in_sizes, int n_in,
                              void* d_out, int out_size, void* d_ws, size_t ws_size,
                              hipStream_t stream) {
  const float* z       = (const float*)d_in[0];
  const int*   chans   = (const int*)d_in[1];
  // d_in[2] = input_time (unused by reference)
  const float* emb     = (const float*)d_in[3];
  const float* phi_w   = (const float*)d_in[4];
  const float* phi_b   = (const float*)d_in[5];
  const float* cs_in   = (const float*)d_in[6];

  float* out      = (float*)d_out;
  float* outZ     = out;                        // 16777216
  float* outLoss  = out + ZQ_ELEMS;             // 1
  float* outIdx   = out + ZQ_ELEMS + 1;         // 1024*85
  float* outClust = out + ZQ_ELEMS + 1 + (size_t)L_ * IDX_PER_L;  // 8192

  // workspace carve-up (f32 region then 32B-aligned bf16/int region)
  float* W = (float*)d_ws;
  size_t fo = 0;
  float* zLND   = W + fo; fo += (size_t)ZQ_ELEMS;
  float* frest  = W + fo; fo += (size_t)ZQ_ELEMS;
  float* fhat   = W + fo; fo += (size_t)ZQ_ELEMS;
  float* hbuf   = W + fo; fo += (size_t)ZQ_ELEMS;
  float* en     = W + fo; fo += (size_t)V_ * D_;
  float* counts = W + fo; fo += (size_t)V_;
  float* lossA  = W + fo; fo += 8;
  char* Bp = (char*)(W + fo);
  Bp = (char*)(((uintptr_t)Bp + 31) & ~(uintptr_t)31);
  __bf16* qbuf   = (__bf16*)Bp; Bp += (size_t)NPOS * D_ * 2;                 // max rows 65536
  __bf16* ebfrag = (__bf16*)Bp; Bp += (size_t)NTILES_EB * KTILES_EB * 512 * 2;
  __bf16* wfrag  = (__bf16*)Bp; Bp += (size_t)NTILES_W * KTILES_W * 512 * 2;
  int*    idxbuf = (int*)Bp;    Bp += (size_t)NPOS * 4;
  (void)ws_size; (void)in_sizes; (void)n_in; (void)out_size;

  hipMemsetAsync(fhat,   0, (size_t)ZQ_ELEMS * 4, stream);
  hipMemsetAsync(counts, 0, (size_t)V_ * 4,       stream);
  hipMemsetAsync(lossA,  0, 32,                   stream);

  k_norm_embed  <<<V_, 256, 0, stream>>>(emb, en);
  k_pack_eb     <<<NTILES_EB * KTILES_EB, 256, 0, stream>>>(en, ebfrag);
  k_pack_w      <<<NTILES_W * KTILES_W, 256, 0, stream>>>(phi_w, wfrag);
  k_norm_scatter<<<B_ * 1024, 256, 0, stream>>>(z, chans, zLND, frest);

  const int gs[4]   = {1, 4, 16, 64};
  const int soff[4] = {0, 1, 5, 21};
  for (int s = 0; s < 4; ++s) {
    int g = gs[s], sub = C_ / g, rows = L_ * g;
    k_group_mean_pack<<<rows, 256, 0, stream>>>(frest, qbuf, g, sub);
    k_argmax        <<<rows / 32, 256, 0, stream>>>(qbuf, ebfrag, idxbuf, counts,
                                                    outIdx, g, soff[s]);
    k_build_h       <<<NPOS, 256, 0, stream>>>(idxbuf, en, hbuf, g, sub);
    k_conv_update   <<<L_ * 2, 256, 0, stream>>>(hbuf, wfrag, phi_b, zLND,
                                                 fhat, frest, lossA);
  }

  k_gather <<<B_ * 1024, 256, 0, stream>>>(fhat, chans, outZ);
  k_cluster<<<V_ / 256, 256, 0, stream>>>(cs_in, counts, outClust);
  k_loss   <<<1, 1, 0, stream>>>(lossA, outLoss);
}
